// Attention_86809878987230
// MI455X (gfx1250) — matrix operand order repaired and checked by reading, not compiled, not run on hardware
//
#include <hip/hip_runtime.h>
#include <hip/hip_bf16.h>
#include <math.h>

// CDNA5 / gfx1250 wave32 WMMA types
typedef __attribute__((ext_vector_type(16))) __bf16 v16bf;
typedef __attribute__((ext_vector_type(8)))  float  v8f;
typedef __attribute__((ext_vector_type(4)))  float  v4f;   // for NT stores

#define BATCH   4
#define HEADS   16
#define SEQ     2048
#define HDIM    64
#define QTILE   16
#define NEG_INF (-1000000000.0f)

__device__ __forceinline__ float4 ld4(const float* p) { return *(const float4*)p; }

__device__ __forceinline__ void cvt4(v16bf& d, int base, float4 x) {
    d[base + 0] = (__bf16)x.x;
    d[base + 1] = (__bf16)x.y;
    d[base + 2] = (__bf16)x.z;
    d[base + 3] = (__bf16)x.w;
}

// One workgroup = one (b,h) pair + one 16-row q tile. 8 wave32 waves.
// Full 16x2048 score row resident in LDS (128KB; CDNA5 has 320KB/WGP).
__global__ __launch_bounds__(256) void attn_fused_kernel(
    const float* __restrict__ q,
    const float* __restrict__ k,
    const float* __restrict__ v,
    const int*   __restrict__ mask,
    float*       __restrict__ out,
    float*       __restrict__ attn)
{
    __shared__ float sc[QTILE][SEQ];        // scores -> exp -> normalized P
    __shared__ float red[QTILE][16];        // per-segment reduction scratch
    __shared__ float rowstat[QTILE];        // row max, then 1/row sum
    __shared__ float partial[QTILE][HDIM];  // PV partial from key-half 1

    const int bh   = blockIdx.x / (SEQ / QTILE);
    const int qt   = blockIdx.x % (SEQ / QTILE);
    const int q0   = qt * QTILE;
    const int tid  = threadIdx.x;
    const int wave = tid >> 5;    // 0..7
    const int lane = tid & 31;
    const int mrow = lane & 15;   // A row / B,C column index for this lane
    const int half = lane >> 4;   // lane group

    const float scale = 0.125f;   // 1/sqrt(64)
    const size_t base_qkv = (size_t)bh * SEQ * HDIM;

    // ---- Q A-fragments: 16x32 bf16 x2 chunks. Per lane, two contiguous
    //      8-float runs per chunk: d = 32c + 8*half + t and +16.
    v16bf aq[2];
    {
        const float* qrow = q + base_qkv + (size_t)(q0 + mrow) * HDIM;
#pragma unroll
        for (int c = 0; c < 2; ++c) {
            const float* p0 = qrow + 32 * c + 8 * half;
            cvt4(aq[c], 0,  ld4(p0 + 0));
            cvt4(aq[c], 4,  ld4(p0 + 4));
            cvt4(aq[c], 8,  ld4(p0 + 16));
            cvt4(aq[c], 12, ld4(p0 + 20));
        }
    }

    // ---- QK^T: wave w handles k-tiles w, w+8, ... (16 tiles each) ----
    // B fragment chunk c is two 8-float runs of K[key][.]:
    //   d = 32c + 8*half + t and +16, t = 0..7  -> 4x float4 loads.
    for (int kt = wave; kt < SEQ / 16; kt += 8) {
        const float* krow = k + base_qkv + (size_t)(kt * 16 + mrow) * HDIM;
        v8f acc = {};
#pragma unroll
        for (int c = 0; c < 2; ++c) {
            v16bf bk;
            const float* kp = krow + 32 * c + 8 * half;
            cvt4(bk, 0,  ld4(kp + 0));
            cvt4(bk, 4,  ld4(kp + 4));
            cvt4(bk, 8,  ld4(kp + 16));
            cvt4(bk, 12, ld4(kp + 20));
            acc = __builtin_amdgcn_wmma_f32_16x16x32_bf16(
                false, aq[c], false, bk, (short)0, acc, false, false);
        }
        // C layout: VGPR r -> row (r + 8*half), column = mrow
        const int key = kt * 16 + mrow;
        const int mk  = mask[key];
#pragma unroll
        for (int r = 0; r < 8; ++r) {
            sc[r + 8 * half][key] = mk ? acc[r] * scale : NEG_INF;
        }
    }
    __syncthreads();

    // ---- row softmax over 2048 keys: 16 threads per row, 128 cols each --
    const int srow = tid >> 4;   // 0..15
    const int seg  = tid & 15;   // 0..15
    {
        float m = NEG_INF;
        const float* p = &sc[srow][seg * 128];
        for (int c = 0; c < 128; c += 4) {
            const float4 x = ld4(p + c);
            m = fmaxf(m, fmaxf(fmaxf(x.x, x.y), fmaxf(x.z, x.w)));
        }
        red[srow][seg] = m;
    }
    __syncthreads();
    if (seg == 0) {
        float m = red[srow][0];
#pragma unroll
        for (int i = 1; i < 16; ++i) m = fmaxf(m, red[srow][i]);
        rowstat[srow] = m;
    }
    __syncthreads();
    {
        const float rm = rowstat[srow];
        float s = 0.0f;
        float* p = &sc[srow][seg * 128];
        for (int c = 0; c < 128; c += 4) {
            float4 x = ld4(p + c);
            x.x = __expf(x.x - rm);
            x.y = __expf(x.y - rm);
            x.z = __expf(x.z - rm);
            x.w = __expf(x.w - rm);
            *(float4*)(p + c) = x;
            s += x.x + x.y + x.z + x.w;
        }
        red[srow][seg] = s;
    }
    __syncthreads();
    if (seg == 0) {
        float s = 0.0f;
#pragma unroll
        for (int i = 0; i < 16; ++i) s += red[srow][i];
        rowstat[srow] = 1.0f / s;
    }
    __syncthreads();

    // ---- normalize in LDS + stream attn out with non-temporal stores ----
    // (attn is write-once; keep the 192MB L2 free for K/V re-use)
    float* attn_bh = attn + (size_t)bh * SEQ * SEQ + (size_t)q0 * SEQ;
    for (int i4 = tid; i4 < (QTILE * SEQ) / 4; i4 += 256) {
        const int idx = i4 * 4;
        const int row = idx >> 11;          // / SEQ
        const int col = idx & (SEQ - 1);
        const float inv = rowstat[row];
        float4 pv = ld4(&sc[row][col]);
        pv.x *= inv; pv.y *= inv; pv.z *= inv; pv.w *= inv;
        *(float4*)&sc[row][col] = pv;
        const v4f nt = { pv.x, pv.y, pv.z, pv.w };
        __builtin_nontemporal_store(nt, (v4f*)(attn_bh + (size_t)row * SEQ + col));
    }
    __syncthreads();

    // ---- P @ V: wave -> (d-tile = wave&3, key-half = wave>>2) ----------
    v8f oacc = {};
    const int d0    = (wave & 3) * 16;
    const int kbase = (wave >> 2) * (SEQ / 2);
    for (int kk = 0; kk < (SEQ / 2) / 32; ++kk) {   // 32 iterations
        const int K0 = kbase + kk * 32;
        v16bf pa, vb;
        // A fragment from LDS: two contiguous 8-float runs per lane
        const float* sp = &sc[mrow][K0 + 8 * half];
        cvt4(pa, 0,  ld4(sp + 0));
        cvt4(pa, 4,  ld4(sp + 4));
        cvt4(pa, 8,  ld4(sp + 16));
        cvt4(pa, 12, ld4(sp + 20));
        // B fragment from V: key-major memory -> strided gather (coalesced
        // across lanes: 16 lanes read 64B-contiguous d-runs per key)
        const float* vp = v + base_qkv + (size_t)(K0 + 8 * half) * HDIM + d0 + mrow;
#pragma unroll
        for (int idx = 0; idx < 16; ++idx) {
            vb[idx] = (__bf16)vp[(idx < 8 ? idx : idx + 8) * HDIM];
        }
        oacc = __builtin_amdgcn_wmma_f32_16x16x32_bf16(
            false, pa, false, vb, (short)0, oacc, false, false);
    }
    if (wave >= 4) {
#pragma unroll
        for (int r = 0; r < 8; ++r) {
            partial[r + 8 * half][d0 + mrow] = oacc[r];
        }
    }
    __syncthreads();
    if (wave < 4) {
#pragma unroll
        for (int r = 0; r < 8; ++r) {
            const int row = r + 8 * half;
            out[base_qkv + (size_t)(q0 + row) * HDIM + d0 + mrow] =
                oacc[r] + partial[row][d0 + mrow];
        }
    }
}

extern "C" void kernel_launch(void* const* d_in, const int* in_sizes, int n_in,
                              void* d_out, int out_size, void* d_ws, size_t ws_size,
                              hipStream_t stream) {
    (void)in_sizes; (void)n_in; (void)out_size; (void)d_ws; (void)ws_size;
    const float* q    = (const float*)d_in[0];
    const float* k    = (const float*)d_in[1];
    const float* v    = (const float*)d_in[2];
    const int*   mask = (const int*)d_in[3];
    float* out  = (float*)d_out;
    float* attn = out + (size_t)BATCH * HEADS * SEQ * HDIM;  // tuple: (output, attn)

    dim3 grid(BATCH * HEADS * (SEQ / QTILE));  // 8192 workgroups
    dim3 block(256);                            // 8 wave32 waves
    attn_fused_kernel<<<grid, block, 0, stream>>>(q, k, v, mask, out, attn);
}
